// RNNEncoder_71846212928315
// MI455X (gfx1250) — compile-verified, matrix operand order checked by reading, not run on hardware
//
#include <hip/hip_runtime.h>

// ---------------------------------------------------------------------------
// Tree-LSTM (perfect binary tree, K=2) for B=512, N=127, D=300 on gfx1250.
// Level-by-level kernels; GEMMs via v_wmma_f32_16x16x32_bf16 (f32 accum).
// ---------------------------------------------------------------------------

typedef __attribute__((ext_vector_type(16))) __bf16          v16bf;
typedef __attribute__((ext_vector_type(8)))  float           v8f;
typedef __attribute__((ext_vector_type(16))) unsigned short  v16u;
typedef __attribute__((ext_vector_type(8)))  unsigned short  v8u;

#define B_       512
#define N_       127
#define D_       300
#define KP       320      // K padded to 10 chunks of 32
#define COLS_IOU 912      // 900 padded to 57*16
#define COLS_F   304      // 300 padded to 19*16

// round-to-nearest-even float -> bf16 bits
__device__ __forceinline__ unsigned short f2bf(float f) {
  unsigned int u = __float_as_uint(f);
  unsigned int r = (u + 0x7FFFu + ((u >> 16) & 1u)) >> 16;
  return (unsigned short)r;
}

// hardware V_TANH_F32 when the builtin exists (gfx1250 trans op), else libm
__device__ __forceinline__ float fast_tanh(float v) {
#if __has_builtin(__builtin_amdgcn_tanhf)
  return __builtin_amdgcn_tanhf(v);
#elif __has_builtin(__builtin_amdgcn_tanh_f32)
  return __builtin_amdgcn_tanh_f32(v);
#else
  return tanhf(v);
#endif
}

// sigmoid(x) = 0.5*tanh(x/2) + 0.5  (single trans op on CDNA5)
__device__ __forceinline__ float sigf(float v) {
  return __builtin_fmaf(0.5f, fast_tanh(0.5f * v), 0.5f);
}

// Load a 16-element bf16 fragment for one lane.  Layout (ISA 7.12.2, 16-bit
// A/B): lane half selects K+0..7 vs K+8..15 within each 16-wide K group;
// elements 0..7 -> K = kb + half*8 + e, elements 8..15 -> K = kb+16+half*8+e.
// With K-contiguous storage (pitch KP) this is two aligned 16-byte loads.
__device__ __forceinline__ v16bf ldfrag(const unsigned short* p) {
  v8u lo = *(const v8u*)(p);
  v8u hi = *(const v8u*)(p + 16);
  v16u r;
#pragma unroll
  for (int e = 0; e < 8; ++e) { r[e] = lo[e]; r[e + 8] = hi[e]; }
  return __builtin_bit_cast(v16bf, r);
}

__device__ __forceinline__ v8f wmma_bf(v16bf a, v16bf b, v8f c) {
  return __builtin_amdgcn_wmma_f32_16x16x32_bf16(
      /*neg_a=*/false, a, /*neg_b=*/false, b,
      /*c_mod=*/(short)0, c, /*reuse_a=*/false, /*reuse_b=*/false);
}

// ---------------------------------------------------------------------------
// Prep: fp32 weights -> bf16, transposed to column-major [col][k], pitch KP,
// zero padded.  Wioux/Wiouh: -> [912][320].  Wfx/Wfh: -> [304][320].
// ---------------------------------------------------------------------------
__global__ void prep_weights_kernel(
    const float* __restrict__ Wioux, const float* __restrict__ Wiouh,
    const float* __restrict__ Wfx,   const float* __restrict__ Wfh,
    unsigned short* __restrict__ WxT, unsigned short* __restrict__ WhT,
    unsigned short* __restrict__ WfxT, unsigned short* __restrict__ WfhT) {
  int i = blockIdx.x * blockDim.x + threadIdx.x;
  if (i >= COLS_IOU * KP) return;
  int col = i / KP;
  int k   = i - col * KP;
  bool inb = (col < 900) && (k < D_);
  WxT[i] = f2bf(inb ? Wioux[k * 900 + col] : 0.0f);
  WhT[i] = f2bf(inb ? Wiouh[k * 900 + col] : 0.0f);
  if (col < COLS_F) {
    bool inb2 = (col < D_) && (k < D_);
    WfxT[i] = f2bf(inb2 ? Wfx[k * D_ + col] : 0.0f);
    WfhT[i] = f2bf(inb2 ? Wfh[k * D_ + col] : 0.0f);
  }
}

// ---------------------------------------------------------------------------
// One tree level.  Each block: 16 rows (row = b*nNodes + j, node = base + j),
// nNodes = 1<<lg.  4 waves split the 19 column tiles of the fused
// i/o/u/fx/f0/f1 GEMMs; LSTM cell epilogue fully in registers.
// ---------------------------------------------------------------------------
__global__ __launch_bounds__(128) void tree_level_kernel(
    const float* __restrict__ x,
    const float* __restrict__ bioux, const float* __restrict__ biouh,
    const float* __restrict__ bfx,   const float* __restrict__ bfh,
    const unsigned short* __restrict__ WxT, const unsigned short* __restrict__ WhT,
    const unsigned short* __restrict__ WfxT, const unsigned short* __restrict__ WfhT,
    float* __restrict__ H, float* __restrict__ C,
    int base, int lg, int hasChildren) {
  __shared__ __align__(16) unsigned short smem[4 * 16 * KP];
  unsigned short* XJ = smem;                // x rows,    bf16 [16][320]
  unsigned short* HS = smem + 16 * KP;      // h0+h1
  unsigned short* H0 = smem + 2 * 16 * KP;  // child0 h
  unsigned short* H1 = smem + 3 * 16 * KP;  // child1 h

  const int tid     = threadIdx.x;
  const int rowTile = blockIdx.x * 16;
  const int nMask   = (1 << lg) - 1;

  // ---- stage inputs (gather children H, mask, convert to bf16, pad K) ----
  for (int i = tid; i < 16 * KP; i += 128) {
    int m = i / KP, k = i - m * KP;
    int row  = rowTile + m;
    int b    = row >> lg;
    int node = base + (row & nMask);
    float xv = 0.0f, h0 = 0.0f, h1 = 0.0f;
    if (k < D_) {
      xv = x[(b * N_ + node) * D_ + k];
      if (hasChildren) {
        h0 = H[(b * N_ + 2 * node + 1) * D_ + k];
        h1 = H[(b * N_ + 2 * node + 2) * D_ + k];
      }
    }
    XJ[i] = f2bf(xv);
    H0[i] = f2bf(h0);
    H1[i] = f2bf(h1);
    HS[i] = f2bf(h0 + h1);
  }
  __syncthreads();

  const int lane = tid & 31;
  const int wave = tid >> 5;
  const int half = lane >> 4;
  const int m    = lane & 15;      // A-fragment row / B-fragment local col
  const int hoff = half * 8;

  for (int t = wave; t < 19; t += 4) {
    const int d   = t * 16;
    const int col = d + m;         // this lane's output column (may be padded)

    // prefetch this wave's next column tile's weight panels into L2/WGP$
    const int dn = d + 64;
    if (dn < COLS_F) {
      const int cn = dn + m;
      __builtin_prefetch(WxT  + (size_t)cn * KP, 0, 1);
      __builtin_prefetch(WxT  + (size_t)(cn + 300) * KP, 0, 1);
      __builtin_prefetch(WxT  + (size_t)(cn + 600) * KP, 0, 1);
      __builtin_prefetch(WhT  + (size_t)cn * KP, 0, 1);
      __builtin_prefetch(WfxT + (size_t)cn * KP, 0, 1);
      __builtin_prefetch(WfhT + (size_t)cn * KP, 0, 1);
    }

    v8f ai = {}, ao = {}, au = {}, afx = {}, af0 = {}, af1 = {};

    for (int kc = 0; kc < 10; ++kc) {
      const int kb = kc * 32;
      v16bf ax  = ldfrag(XJ + m * KP + kb + hoff);
      v16bf b_i = ldfrag(WxT  + (size_t)col * KP + kb + hoff);
      v16bf b_o = ldfrag(WxT  + (size_t)(col + 300) * KP + kb + hoff);
      v16bf b_u = ldfrag(WxT  + (size_t)(col + 600) * KP + kb + hoff);
      v16bf b_f = ldfrag(WfxT + (size_t)col * KP + kb + hoff);
      ai  = wmma_bf(ax, b_i, ai);
      ao  = wmma_bf(ax, b_o, ao);
      au  = wmma_bf(ax, b_u, au);
      afx = wmma_bf(ax, b_f, afx);
      if (hasChildren) {                        // wave-uniform branch
        v16bf ah   = ldfrag(HS + m * KP + kb + hoff);
        v16bf bh_i = ldfrag(WhT + (size_t)col * KP + kb + hoff);
        v16bf bh_o = ldfrag(WhT + (size_t)(col + 300) * KP + kb + hoff);
        v16bf bh_u = ldfrag(WhT + (size_t)(col + 600) * KP + kb + hoff);
        ai = wmma_bf(ah, bh_i, ai);
        ao = wmma_bf(ah, bh_o, ao);
        au = wmma_bf(ah, bh_u, au);
        v16bf a0   = ldfrag(H0 + m * KP + kb + hoff);
        v16bf a1   = ldfrag(H1 + m * KP + kb + hoff);
        v16bf b_fh = ldfrag(WfhT + (size_t)col * KP + kb + hoff);
        af0 = wmma_bf(a0, b_fh, af0);
        af1 = wmma_bf(a1, b_fh, af1);
      }
    }

    // ---- elementwise LSTM cell on this 16x16 patch (registers only) ----
    const bool colOK = (col < D_);
    const int  cb = colOK ? col : (D_ - 1);                 // clamped reads
    const int  co = (col + 300 < 900) ? (col + 300) : 899;
    const int  cu = (col + 600 < 900) ? (col + 600) : 899;
    const float Bi  = bioux[cb] + biouh[cb];
    const float Bo  = bioux[co] + biouh[co];
    const float Bu  = bioux[cu] + biouh[cu];
    const float Bfx = bfx[cb];
    const float Bfh = bfh[cb];

#pragma unroll
    for (int r = 0; r < 8; ++r) {
      const int mr   = r + half * 8;          // D-matrix row for this VGPR
      const int row  = rowTile + mr;
      const int b    = row >> lg;
      const int node = base + (row & nMask);
      float ig  = sigf(ai[r] + Bi);
      float og  = sigf(ao[r] + Bo);
      float ug  = fast_tanh(au[r] + Bu);
      float fxv = afx[r] + Bfx;
      float c   = ig * ug;
      if (hasChildren) {
        float cc0 = 0.0f, cc1 = 0.0f;
        if (colOK) {
          cc0 = C[(b * N_ + 2 * node + 1) * D_ + col];
          cc1 = C[(b * N_ + 2 * node + 2) * D_ + col];
        }
        float f0 = sigf(af0[r] + Bfh + fxv);
        float f1 = sigf(af1[r] + Bfh + fxv);
        c += f0 * cc0 + f1 * cc1;
      }
      float h = og * fast_tanh(c);
      if (colOK) {
        H[(b * N_ + node) * D_ + col] = h;
        C[(b * N_ + node) * D_ + col] = c;
      }
    }
  }
}

// ---------------------------------------------------------------------------
extern "C" void kernel_launch(void* const* d_in, const int* in_sizes, int n_in,
                              void* d_out, int out_size, void* d_ws, size_t ws_size,
                              hipStream_t stream) {
  const float* x     = (const float*)d_in[0];
  const float* Wioux = (const float*)d_in[1];
  const float* bioux = (const float*)d_in[2];
  const float* Wiouh = (const float*)d_in[3];
  const float* biouh = (const float*)d_in[4];
  const float* Wfx   = (const float*)d_in[5];
  const float* bfx   = (const float*)d_in[6];
  const float* Wfh   = (const float*)d_in[7];
  const float* bfh   = (const float*)d_in[8];

  // workspace layout: bf16 transposed weights, then fp32 C state
  unsigned short* WxT  = (unsigned short*)d_ws;
  unsigned short* WhT  = WxT  + (size_t)COLS_IOU * KP;
  unsigned short* WfxT = WhT  + (size_t)COLS_IOU * KP;
  unsigned short* WfhT = WfxT + (size_t)COLS_F * KP;
  float*          Cbuf = (float*)(WfhT + (size_t)COLS_F * KP);

  float* H = (float*)d_out;   // (B, N, D) — final H is the output

  const int prepN = COLS_IOU * KP;
  prep_weights_kernel<<<(prepN + 255) / 256, 256, 0, stream>>>(
      Wioux, Wiouh, Wfx, Wfh, WxT, WhT, WfxT, WfhT);

  // perfect binary tree: level l has 2^l nodes starting at 2^l - 1;
  // leaves (l == 6) have no children.  Deepest level first.
  for (int l = 6; l >= 0; --l) {
    int n     = 1 << l;
    int basei = n - 1;
    int R     = B_ * n;                 // rows this level (multiple of 16)
    tree_level_kernel<<<R / 16, 128, 0, stream>>>(
        x, bioux, biouh, bfx, bfh, WxT, WhT, WfxT, WfhT,
        H, Cbuf, basei, l, (l < 6) ? 1 : 0);
  }
}
